// OuterProductMean_31645319037649
// MI455X (gfx1250) — matrix operand order repaired and checked by reading, not compiled, not run on hardware
//
#include <hip/hip_runtime.h>
#include <hip/hip_bf16.h>

// ---- problem constants (from reference setup_inputs) ----
#define NSEQ 256
#define NRES 384
#define CM   256
#define CH   32
#define CZ   128
#define LN_EPS  1e-5f
#define OPM_EPS 1e-3f

// LDS padding (bf16 elements) to kill bank conflicts on ds_load_b128
#define PADB 40     // B-stage row stride (32 K-values + 8 pad) -> 80B
#define PADO 1032   // O2 row stride (1024 + 8 pad) -> 2064B == 16 mod 256

typedef __bf16 bf16;
typedef __attribute__((ext_vector_type(8)))  float v8f;
typedef __attribute__((ext_vector_type(16))) __bf16 v16bf;
typedef __attribute__((ext_vector_type(8)))  __bf16 v8bf;

static __device__ __forceinline__ bf16 f2bf(float f) {
  unsigned u = __builtin_bit_cast(unsigned, f);
  u += 0x7FFFu + ((u >> 16) & 1u);            // round-to-nearest-even
  unsigned short h = (unsigned short)(u >> 16);
  return __builtin_bit_cast(bf16, h);
}

static __device__ __forceinline__ v16bf combine16(v8bf lo, v8bf hi) {
  v16bf f;
  #pragma unroll
  for (int e = 0; e < 8; ++e) { f[e] = lo[e]; f[8 + e] = hi[e]; }
  return f;
}

// =====================================================================
// Kernel 1: LayerNorm over c_m + both 256->32 projections (+mask), store
// a_t/b_t as bf16 in "contraction-major" layout: [(res*32+ch)][s].
// =====================================================================
__global__ __launch_bounds__(256) void opm_ln_proj(
    const float* __restrict__ m, const float* __restrict__ mask,
    const float* __restrict__ ln_w, const float* __restrict__ ln_b,
    const float* __restrict__ w1, const float* __restrict__ b1,
    const float* __restrict__ w2, const float* __restrict__ b2,
    bf16* __restrict__ a_t, bf16* __restrict__ b_t)
{
  const int row = blockIdx.x;               // 0 .. NSEQ*NRES-1
  const int s = row / NRES, i = row % NRES;
  const int t = threadIdx.x;                // 0..255 == channel k

  __shared__ float xs[CM];
  __shared__ float red[32];
  __shared__ float accs[2][8][CH];

  const float v = m[(size_t)row * CM + t];

  float sum = v, sq = v * v;
  #pragma unroll
  for (int off = 16; off; off >>= 1) {
    sum += __shfl_down(sum, off, 32);
    sq  += __shfl_down(sq,  off, 32);
  }
  const int wave = t >> 5, lane = t & 31;
  if (lane == 0) { red[wave] = sum; red[8 + wave] = sq; }
  __syncthreads();
  if (t == 0) {
    float S = 0.f, Q = 0.f;
    #pragma unroll
    for (int w = 0; w < 8; ++w) { S += red[w]; Q += red[8 + w]; }
    const float mu = S * (1.0f / CM);
    red[16] = mu;
    red[17] = rsqrtf(Q * (1.0f / CM) - mu * mu + LN_EPS);
  }
  __syncthreads();
  const float mu = red[16], rstd = red[17];
  xs[t] = (v - mu) * rstd * ln_w[t] + ln_b[t];
  __syncthreads();

  const int c = t & 31, chunk = t >> 5;
  float pa = 0.f, pb = 0.f;
  #pragma unroll 4
  for (int k = chunk * 32; k < chunk * 32 + 32; ++k) {
    const float xv = xs[k];
    pa += xv * w1[k * CH + c];
    pb += xv * w2[k * CH + c];
  }
  accs[0][chunk][c] = pa;
  accs[1][chunk][c] = pb;
  __syncthreads();

  if (t < 64) {
    const int which = t >> 5, cc = t & 31;
    float acc = 0.f;
    #pragma unroll
    for (int w = 0; w < 8; ++w) acc += accs[which][w][cc];
    const float msk = mask[s * NRES + i];
    const float val = (acc + (which ? b2[cc] : b1[cc])) * msk;
    bf16* dst = which ? b_t : a_t;
    dst[((size_t)i * CH + cc) * NSEQ + s] = f2bf(val);
  }
}

// =====================================================================
// Kernel 2: wo[1024,128] f32 -> wo_t[z][ce] bf16.
// =====================================================================
__global__ __launch_bounds__(256) void opm_wo_prep(
    const float* __restrict__ wo, bf16* __restrict__ wo_t)
{
  const int idx = blockIdx.x * 256 + threadIdx.x;   // 0..131071
  const int z  = idx >> 10;
  const int ce = idx & 1023;
  wo_t[(size_t)z * 1024 + ce] = f2bf(wo[(size_t)ce * CZ + z]);
}

// =====================================================================
// Kernel 3: fused outer-product-mean, software-pipelined.
// =====================================================================
__global__ __launch_bounds__(256) void opm_outer_wmma(
    const bf16* __restrict__ a_t, const bf16* __restrict__ b_t,
    const bf16* __restrict__ wo_t, const float* __restrict__ bo,
    const float* __restrict__ mask, float* __restrict__ out)
{
  const int it0 = blockIdx.x;          // 0..95 (i tile of 4)
  const int jt0 = blockIdx.y;          // 0..95 (j tile of 4)
  const int t = threadIdx.x;
  const int wave = t >> 5;
  const int lane = t & 31;
  const int half = lane >> 4;          // 0: lanes 0-15, 1: lanes 16-31
  const int r    = lane & 15;

  __shared__ alignas(16) bf16 Bs[2][128 * PADB];   // double-buffered B K-chunks
  __shared__ alignas(16) bf16 O2[16 * PADO];       // 16 pairs x (c*32+e), padded
  __shared__ float norms[16];

  // ---- mask norm per pair: norm = sum_s mask[s,i]*mask[s,j] ----
  #pragma unroll
  for (int pp = 0; pp < 2; ++pp) {
    const int p  = wave * 2 + pp;
    const int gi = it0 * 4 + (p >> 2);
    const int gj = jt0 * 4 + (p & 3);
    float nacc = 0.f;
    for (int s = lane; s < NSEQ; s += 32)
      nacc += mask[s * NRES + gi] * mask[s * NRES + gj];
    #pragma unroll
    for (int off = 16; off; off >>= 1) nacc += __shfl_down(nacc, off, 32);
    if (lane == 0) norms[p] = nacc;
  }

  // ---- cooperative B-panel stager: thread t owns 32B of each 8KB chunk ----
  const int bcol  = t >> 1;            // 0..127
  const int bhalf = t & 1;
  const bf16* Bsrc = b_t + (size_t)(jt0 * 128 + bcol) * NSEQ + bhalf * 16;
  const int bofs = bcol * PADB + bhalf * 16;

  // preload K-chunk 0 into buffer 0
  {
    v8bf p0 = *(const v8bf*)(Bsrc);
    v8bf p1 = *(const v8bf*)(Bsrc + 8);
    *(v8bf*)(&Bs[0][bofs])     = p0;
    *(v8bf*)(&Bs[0][bofs + 8]) = p1;
  }
  __syncthreads();

  // ---- stage 1: O[(i,c),(j,e)] = sum_s a*b. Wave w owns M-tile w ----
  v8f acc[8];
  #pragma unroll
  for (int n = 0; n < 8; ++n)
    #pragma unroll
    for (int k = 0; k < 8; ++k) acc[n][k] = 0.f;

  const bf16* Abase = a_t + (size_t)(it0 * 128 + wave * 16 + r) * NSEQ;

  // A fragment pipeline registers (step 0 preloaded)
  v8bf alo = *(const v8bf*)(Abase + half * 16);
  v8bf ahi = *(const v8bf*)(Abase + half * 16 + 8);

  #pragma unroll
  for (int step = 0; step < 8; ++step) {
    const int ks = step * 32;

    // prefetch next B chunk (global->regs) and next A fragment
    v8bf n0, n1, nalo, nahi;
    if (step < 7) {
      n0   = *(const v8bf*)(Bsrc + ks + 32);
      n1   = *(const v8bf*)(Bsrc + ks + 32 + 8);
      nalo = *(const v8bf*)(Abase + ks + 32 + half * 16);
      nahi = *(const v8bf*)(Abase + ks + 32 + half * 16 + 8);
    }

    const v16bf af = combine16(alo, ahi);

    // gather ALL 8 B fragments first (distinct regs -> one dscnt wait),
    // then fire 8 back-to-back WMMAs sharing A.
    const bf16* Bbuf = &Bs[step & 1][0];
    v16bf bfrag[8];
    #pragma unroll
    for (int nt = 0; nt < 8; ++nt) {
      const bf16* bp = Bbuf + (nt * 16 + r) * PADB + half * 16;
      bfrag[nt] = combine16(*(const v8bf*)(bp), *(const v8bf*)(bp + 8));
    }
    #pragma unroll
    for (int nt = 0; nt < 8; ++nt) {
      acc[nt] = __builtin_amdgcn_wmma_f32_16x16x32_bf16(
          false, af, false, bfrag[nt], (short)0, acc[nt], false, false);
    }

    if (step < 7) {
      bf16* d = &Bs[(step + 1) & 1][bofs];
      *(v8bf*)(d)     = n0;
      *(v8bf*)(d + 8) = n1;
      alo = nalo; ahi = nahi;
    }
    __syncthreads();
  }

  // C/D layout: VGPR v, lane(half,r): M = v + half*8, N = r (within tile)
  #pragma unroll
  for (int nt = 0; nt < 8; ++nt) {
    #pragma unroll
    for (int v = 0; v < 8; ++v) {
      const int M = wave * 16 + v + half * 8;   // 0..127 -> (i_local, c)
      const int N = nt * 16 + r;                // 0..127 -> (j_local, e)
      const int iL = M >> 5, cc = M & 31;
      const int jL = N >> 5, ee = N & 31;
      O2[(iL * 4 + jL) * PADO + cc * 32 + ee] = f2bf(acc[nt][v]);
    }
  }
  __syncthreads();

  // ---- stage 2: out[pair, z] = O2[pair, :] @ wo_t^T ; wave w owns z-tile w
  v8f acc2;
  #pragma unroll
  for (int k = 0; k < 8; ++k) acc2[k] = 0.f;

  const int z = wave * 16 + r;
  const bf16* Wbase = wo_t + (size_t)z * 1024;
  const bf16* Obase = &O2[r * PADO];

  // double-buffered operand regs: slot kk&1 holds iteration kk's fragments
  v8bf AL[2], AH[2], BL[2], BH[2];
  AL[0] = *(const v8bf*)(Obase + half * 16);
  AH[0] = *(const v8bf*)(Obase + half * 16 + 8);
  BL[0] = *(const v8bf*)(Wbase + half * 16);
  BH[0] = *(const v8bf*)(Wbase + half * 16 + 8);

  #pragma unroll
  for (int kk = 0; kk < 32; ++kk) {
    const int cur = kk & 1, nxt = cur ^ 1;
    if (kk < 31) {
      const int ks = (kk + 1) * 32;
      AL[nxt] = *(const v8bf*)(Obase + ks + half * 16);
      AH[nxt] = *(const v8bf*)(Obase + ks + half * 16 + 8);
      BL[nxt] = *(const v8bf*)(Wbase + ks + half * 16);
      BH[nxt] = *(const v8bf*)(Wbase + ks + half * 16 + 8);
    }
    const v16bf af  = combine16(AL[cur], AH[cur]);
    const v16bf bfm = combine16(BL[cur], BH[cur]);
    acc2 = __builtin_amdgcn_wmma_f32_16x16x32_bf16(
        false, af, false, bfm, (short)0, acc2, false, false);
  }

  // epilogue: (acc + bo[z]) / (eps + norm[i,j])
  const float bz = bo[z];
  #pragma unroll
  for (int v = 0; v < 8; ++v) {
    const int p  = v + half * 8;               // pair index 0..15
    const int gi = it0 * 4 + (p >> 2);
    const int gj = jt0 * 4 + (p & 3);
    const float scale = 1.0f / (OPM_EPS + norms[p]);
    out[((size_t)gi * NRES + gj) * CZ + z] = (acc2[v] + bz) * scale;
  }
}

// =====================================================================
// Host launcher
// =====================================================================
extern "C" void kernel_launch(void* const* d_in, const int* in_sizes, int n_in,
                              void* d_out, int out_size, void* d_ws, size_t ws_size,
                              hipStream_t stream) {
  const float* m    = (const float*)d_in[0];
  const float* mask = (const float*)d_in[1];
  const float* ln_w = (const float*)d_in[2];
  const float* ln_b = (const float*)d_in[3];
  const float* w1   = (const float*)d_in[4];
  const float* b1   = (const float*)d_in[5];
  const float* w2   = (const float*)d_in[6];
  const float* b2   = (const float*)d_in[7];
  const float* wo   = (const float*)d_in[8];
  const float* bo   = (const float*)d_in[9];
  float* out = (float*)d_out;

  const size_t AB_BYTES = (size_t)NRES * CH * NSEQ * sizeof(bf16);  // 6 MB each
  char* ws = (char*)d_ws;
  bf16* a_t  = (bf16*)ws;
  bf16* b_t  = (bf16*)(ws + AB_BYTES);
  bf16* wo_t = (bf16*)(ws + 2 * AB_BYTES);

  opm_ln_proj<<<NSEQ * NRES, 256, 0, stream>>>(m, mask, ln_w, ln_b,
                                               w1, b1, w2, b2, a_t, b_t);
  opm_wo_prep<<<(CH * CH * CZ) / 256, 256, 0, stream>>>(wo, wo_t);
  opm_outer_wmma<<<dim3(NRES / 4, NRES / 4), 256, 0, stream>>>(
      a_t, b_t, wo_t, bo, mask, out);
}
